// SelfAttention_67216238182947
// MI455X (gfx1250) — matrix operand order repaired and checked by reading, not compiled, not run on hardware
//
#include <hip/hip_runtime.h>
#include <hip/hip_bf16.h>

typedef _Float16 v16h __attribute__((ext_vector_type(16)));
typedef _Float16 v8h  __attribute__((ext_vector_type(8)));
typedef float    v8f  __attribute__((ext_vector_type(8)));
typedef int      v4i  __attribute__((ext_vector_type(4)));

#define BATCH    2
#define NCH      256
#define NTOK     4096
#define NHEAD    8
#define HDIM     32
#define QKV_ROWS 768
#define INV_SCALE 0.17677669529663687f
#define LN_EPS    1e-5f

// ---------------------------------------------------------------------------
// CDNA5 async global->LDS copy (ASYNCcnt path), with portable fallback.
// Builtin signature (from hipcc diagnostic): (int4 AS1*, int4 AS3*, Ii, Ii).
// ---------------------------------------------------------------------------
#if __has_builtin(__builtin_amdgcn_global_load_async_to_lds_b128)
#define HAVE_ASYNC_LDS 1
#define ASYNC_CP16(lds_p, g_p)                                                   \
  __builtin_amdgcn_global_load_async_to_lds_b128(                                \
      (__attribute__((address_space(1))) v4i*)(__attribute__((address_space(1))) \
           void*)(void*)const_cast<_Float16*>(g_p),                              \
      (__attribute__((address_space(3))) v4i*)(__attribute__((address_space(3))) \
           void*)(void*)(lds_p),                                                 \
      0, 0)
#else
#define HAVE_ASYNC_LDS 0
#define ASYNC_CP16(lds_p, g_p) (*(v8h*)(lds_p) = *(const v8h*)(g_p))
#endif

#if HAVE_ASYNC_LDS
#if __has_builtin(__builtin_amdgcn_s_wait_asynccnt)
#define WAIT_ASYNC(n) __builtin_amdgcn_s_wait_asynccnt(n)
#else
#define WAIT_ASYNC(n) asm volatile("s_wait_asynccnt %0" ::"i"(n) : "memory")
#endif
#else
#define WAIT_ASYNC(n) ((void)0)
#endif

// Build a v16h from two 8-half chunks
static __device__ __forceinline__ v16h cat16(v8h a, v8h b) {
  v16h r;
#pragma unroll
  for (int i = 0; i < 8; ++i) { r[i] = a[i]; r[i + 8] = b[i]; }
  return r;
}

// A-operand fragment: row = lane%16, K chunks {8*hi..8*hi+7} and {16+8*hi..}
static __device__ __forceinline__ v16h load_a(const _Float16* row_k0, int hi) {
  const _Float16* p = row_k0 + 8 * hi;
  return cat16(*(const v8h*)p, *(const v8h*)(p + 16));
}
// B-operand fragment: col = lane%16, K chunks {8*hi..8*hi+7} and {16+8*hi..}
static __device__ __forceinline__ v16h load_b(const _Float16* row_k0, int hi) {
  const _Float16* p = row_k0 + 8 * hi;
  return cat16(*(const v8h*)p, *(const v8h*)(p + 16));
}

#define WMMA(A, B, C) \
  __builtin_amdgcn_wmma_f32_16x16x32_f16(false, (A), false, (B), (short)0, (C), false, false)

// ---------------------------------------------------------------------------
// Kernel 1: convert fp32 weights to f16 in workspace
// ---------------------------------------------------------------------------
__global__ __launch_bounds__(256) void wcvt_kernel(const float* __restrict__ qkvw,
                                                   const float* __restrict__ projw,
                                                   _Float16* __restrict__ qkvw_h,
                                                   _Float16* __restrict__ projw_h) {
  const int stride = gridDim.x * blockDim.x;
  for (int i = blockIdx.x * blockDim.x + threadIdx.x; i < QKV_ROWS * NCH; i += stride)
    qkvw_h[i] = (_Float16)qkvw[i];
  for (int i = blockIdx.x * blockDim.x + threadIdx.x; i < NCH * NCH; i += stride)
    projw_h[i] = (_Float16)projw[i];
}

// ---------------------------------------------------------------------------
// Kernel 2: LayerNorm over channels. One lane per token (coalesced reads of
// the (B,C,N) input). Output x_norm is (B*N, C) row-major f16.
// ---------------------------------------------------------------------------
__global__ __launch_bounds__(256) void ln_kernel(const float* __restrict__ x,
                                                 const float* __restrict__ g,
                                                 const float* __restrict__ bta,
                                                 _Float16* __restrict__ xn) {
  const int wave = threadIdx.x >> 5, lane = threadIdx.x & 31;
  const int row  = (blockIdx.x * 8 + wave) * 32 + lane;   // [0, B*N)
  const int b = row >> 12, tok = row & (NTOK - 1);
  const float* xb = x + (size_t)b * NCH * NTOK + tok;
  float s = 0.f, ss = 0.f;
  for (int c = 0; c < NCH; ++c) {
    float v = xb[(size_t)c * NTOK];
    s += v; ss += v * v;
  }
  const float mu  = s * (1.0f / NCH);
  const float var = ss * (1.0f / NCH) - mu * mu;
  const float r   = rsqrtf(var + LN_EPS);
  _Float16* orow = xn + (size_t)row * NCH;
  for (int c = 0; c < NCH; ++c) {
    float v = xb[(size_t)c * NTOK];
    orow[c] = (_Float16)((v - mu) * r * g[c] + bta[c]);
  }
}

// ---------------------------------------------------------------------------
// Kernel 3: QKV GEMM (M=8192, N=768, K=256). 32x32 output tile per wave
// (2 A-frags x 2 B-frags -> 4 WMMAs per K-step). Scatter into Q (prescaled
// 1/sqrt(d), NxD), K (NxD), V^T (DxN) per head.
// ---------------------------------------------------------------------------
__global__ __launch_bounds__(256) void qkv_kernel(const _Float16* __restrict__ xn,
                                                  const _Float16* __restrict__ wq,
                                                  const float* __restrict__ qkvb,
                                                  _Float16* __restrict__ qh,
                                                  _Float16* __restrict__ kh,
                                                  _Float16* __restrict__ vth) {
  const int wave = threadIdx.x >> 5, lane = threadIdx.x & 31;
  const int lo = lane & 15, hi = lane >> 4;
  const int tid = blockIdx.x * 8 + wave;            // 6144 tiles = 256 x 24
  const int mt = tid / 24, nt = tid % 24;           // 32-wide tiles
  v8f acc00 = {}, acc01 = {}, acc10 = {}, acc11 = {};
  const _Float16* ar0 = xn + (size_t)(mt * 32 + lo) * NCH;
  const _Float16* ar1 = xn + (size_t)(mt * 32 + 16 + lo) * NCH;
  const _Float16* br0 = wq + (size_t)(nt * 32 + lo) * NCH;
  const _Float16* br1 = wq + (size_t)(nt * 32 + 16 + lo) * NCH;
#pragma unroll
  for (int kt = 0; kt < 8; ++kt) {
    const int k0 = kt * 32;
    v16h a0 = load_a(ar0 + k0, hi);
    v16h a1 = load_a(ar1 + k0, hi);
    v16h b0 = load_b(br0 + k0, hi);
    v16h b1 = load_b(br1 + k0, hi);
    acc00 = WMMA(a0, b0, acc00);
    acc01 = WMMA(a0, b1, acc01);
    acc10 = WMMA(a1, b0, acc10);
    acc11 = WMMA(a1, b1, acc11);
  }
  v8f accs[2][2] = {{acc00, acc01}, {acc10, acc11}};
#pragma unroll
  for (int ms = 0; ms < 2; ++ms) {
#pragma unroll
    for (int ns = 0; ns < 2; ++ns) {
      const int n = nt * 32 + ns * 16 + lo;   // output feature [0,768)
      const int sel = n >> 8;                  // 0:q 1:k 2:v (uniform per subtile)
      const int c = n & 255;
      const int head = c >> 5;
      const int dim = c & 31;
      const float bias = qkvb[n];
#pragma unroll
      for (int j = 0; j < 8; ++j) {
        const int mrow = mt * 32 + ms * 16 + j + 8 * hi;
        const int b = mrow >> 12, tok = mrow & (NTOK - 1);
        const int bh = b * NHEAD + head;
        const float v = accs[ms][ns][j] + bias;
        if (sel == 0)
          qh[((size_t)(bh * NTOK + tok)) * HDIM + dim] = (_Float16)(v * INV_SCALE);
        else if (sel == 1)
          kh[((size_t)(bh * NTOK + tok)) * HDIM + dim] = (_Float16)v;
        else
          vth[((size_t)(bh * HDIM + dim)) * NTOK + tok] = (_Float16)v;
      }
    }
  }
}

// ---------------------------------------------------------------------------
// Kernel 4: flash-style attention with block-level K/V reuse.
// All 8 waves of a block work on the SAME (b,head); the block cooperatively
// double-buffers 64-key K/V tiles into LDS via async global->LDS copies
// (2 b128 copies per thread per stage -> s_wait_asynccnt 2 retires the
// in-flight stage). Each wave owns 32 queries: per 32-key sub-block it does
// 4 WMMA for S, online softmax, P relayout via per-wave LDS, 4 WMMA for PV.
// ---------------------------------------------------------------------------
__global__ __launch_bounds__(256) void attn_kernel(const _Float16* __restrict__ qh,
                                                   const _Float16* __restrict__ kh,
                                                   const _Float16* __restrict__ vth,
                                                   _Float16* __restrict__ attn_h) {
  __shared__ __align__(16) _Float16 kbuf[2][64 * 32];  // 64 keys x 32 dims, dbl buf
  __shared__ __align__(16) _Float16 vbuf[2][32 * 64];  // 32 dims x 64 keys, dbl buf
  __shared__ __align__(16) _Float16 pbuf[8][32 * 32];  // per-wave P scratch

  const int wave = threadIdx.x >> 5, lane = threadIdx.x & 31;
  const int lo = lane & 15, hi = lane >> 4;
  const int bh   = blockIdx.x >> 4;                  // 16 (b,head) pairs
  const int qblk = blockIdx.x & 15;                  // 16 blocks per bh
  const int qt0  = (qblk * 8 + wave) * 32;           // this wave's 32 queries

  const _Float16* kbh = kh + (size_t)bh * NTOK * HDIM;
  const _Float16* vbh = vth + (size_t)bh * HDIM * NTOK;

  // Cooperative stage of one 64-key tile (8 KB) into LDS buffer `buf`.
  auto stage = [&](int k0, int buf) {
    const int t = threadIdx.x;                       // 0..255
    // K tile: 64 rows x 64B, contiguous 4 KB -> one b128 per thread
    ASYNC_CP16(&kbuf[buf][t * 8], kbh + (size_t)k0 * HDIM + t * 8);
    // V^T tile: 32 rows (dims) x 128B -> one b128 per thread
    const int dim = t >> 3, seg = t & 7;
    ASYNC_CP16(&vbuf[buf][dim * 64 + seg * 8],
               vbh + (size_t)dim * NTOK + k0 + seg * 8);
  };

  // Persistent Q fragments (2 x 16 queries, d=32 = one A frag each)
  v16h qa[2];
#pragma unroll
  for (int ms = 0; ms < 2; ++ms) {
    const _Float16* qp = qh + ((size_t)(bh * NTOK + qt0 + ms * 16 + lo)) * HDIM;
    qa[ms] = load_a(qp, hi);
  }

  v8f o[2][2] = {};
  float m[2][8], l[2][8];
#pragma unroll
  for (int ms = 0; ms < 2; ++ms)
#pragma unroll
    for (int j = 0; j < 8; ++j) { m[ms][j] = -__builtin_inff(); l[ms][j] = 0.f; }

  _Float16* pl = pbuf[wave];
  const int NIT = NTOK / 64;   // 64 iterations of 64 keys

  stage(0, 0);
  for (int it = 0; it < NIT; ++it) {
    const int cur = it & 1;
    if (it + 1 < NIT) {
      stage((it + 1) * 64, cur ^ 1);   // prefetch next tile into other buffer
      WAIT_ASYNC(2);                   // retire current tile (in-order)
    } else {
      WAIT_ASYNC(0);
    }
    __syncthreads();

    const _Float16* kt = kbuf[cur];
    const _Float16* vt = vbuf[cur];
#pragma unroll
    for (int ks = 0; ks < 64; ks += 32) {
      // K rows as B-operand columns (from LDS), shared by both query subtiles
      v16h kb0 = load_b(kt + (size_t)(ks + lo) * HDIM, hi);
      v16h kb1 = load_b(kt + (size_t)(ks + 16 + lo) * HDIM, hi);
      v8f s[2][2] = {};
#pragma unroll
      for (int ms = 0; ms < 2; ++ms) {
        s[ms][0] = WMMA(qa[ms], kb0, s[ms][0]);
        s[ms][1] = WMMA(qa[ms], kb1, s[ms][1]);
      }
#pragma unroll
      for (int ms = 0; ms < 2; ++ms) {
#pragma unroll
        for (int j = 0; j < 8; ++j) {
          const float a = s[ms][0][j], b = s[ms][1][j];
          float mx = fmaxf(a, b);
#pragma unroll
          for (int off = 8; off >= 1; off >>= 1) mx = fmaxf(mx, __shfl_xor(mx, off));
          const float mn = fmaxf(m[ms][j], mx);
          const float corr = __expf(m[ms][j] - mn);
          const float p0 = __expf(a - mn);
          const float p1 = __expf(b - mn);
          float ps = p0 + p1;
#pragma unroll
          for (int off = 8; off >= 1; off >>= 1) ps += __shfl_xor(ps, off);
          l[ms][j] = l[ms][j] * corr + ps;
          m[ms][j] = mn;
          o[ms][0][j] *= corr;
          o[ms][1][j] *= corr;
          const int row = ms * 16 + j + 8 * hi;    // C-layout row for this half
          pl[row * 32 + lo]      = (_Float16)p0;
          pl[row * 32 + 16 + lo] = (_Float16)p1;
        }
      }
      asm volatile("s_wait_dscnt 0x0" ::: "memory");
      // V^T rows (dims) as B-operand columns over this 32-key block (from LDS)
      v16h vb0 = load_b(vt + (size_t)lo * 64 + ks, hi);
      v16h vb1 = load_b(vt + (size_t)(16 + lo) * 64 + ks, hi);
#pragma unroll
      for (int ms = 0; ms < 2; ++ms) {
        v16h pa = load_a(pl + (size_t)(ms * 16 + lo) * 32, hi);
        o[ms][0] = WMMA(pa, vb0, o[ms][0]);
        o[ms][1] = WMMA(pa, vb1, o[ms][1]);
      }
    }
    __syncthreads();
  }

  const int b = bh >> 3, h = bh & 7;
#pragma unroll
  for (int ms = 0; ms < 2; ++ms) {
#pragma unroll
    for (int j = 0; j < 8; ++j) {
      const int tok = qt0 + ms * 16 + j + 8 * hi;
      const float inv = 1.0f / l[ms][j];
      const size_t base = ((size_t)(b * NTOK + tok)) * NCH + h * HDIM;
      attn_h[base + lo]      = (_Float16)(o[ms][0][j] * inv);
      attn_h[base + 16 + lo] = (_Float16)(o[ms][1][j] * inv);
    }
  }
}

// ---------------------------------------------------------------------------
// Kernel 5: output projection (8192x256x256) + bias + residual, 32x32 tile
// per wave, writing the (B, C, N) output directly (8 consecutive floats/lane).
// ---------------------------------------------------------------------------
__global__ __launch_bounds__(256) void proj_kernel(const _Float16* __restrict__ attn_h,
                                                   const _Float16* __restrict__ wp,
                                                   const float* __restrict__ pb,
                                                   const float* __restrict__ x,
                                                   float* __restrict__ out) {
  const int wave = threadIdx.x >> 5, lane = threadIdx.x & 31;
  const int lo = lane & 15, hi = lane >> 4;
  const int tid = blockIdx.x * 8 + wave;   // 2048 tiles = 256 x 8
  const int mt = tid >> 3, nt = tid & 7;   // 32-wide tiles
  v8f acc00 = {}, acc01 = {}, acc10 = {}, acc11 = {};
  const _Float16* ar0 = attn_h + (size_t)(mt * 32 + lo) * NCH;
  const _Float16* ar1 = attn_h + (size_t)(mt * 32 + 16 + lo) * NCH;
  const _Float16* br0 = wp + (size_t)(nt * 32 + lo) * NCH;
  const _Float16* br1 = wp + (size_t)(nt * 32 + 16 + lo) * NCH;
#pragma unroll
  for (int kt = 0; kt < 8; ++kt) {
    const int k0 = kt * 32;
    v16h a0 = load_a(ar0 + k0, hi);
    v16h a1 = load_a(ar1 + k0, hi);
    v16h b0 = load_b(br0 + k0, hi);
    v16h b1 = load_b(br1 + k0, hi);
    acc00 = WMMA(a0, b0, acc00);
    acc01 = WMMA(a0, b1, acc01);
    acc10 = WMMA(a1, b0, acc10);
    acc11 = WMMA(a1, b1, acc11);
  }
  v8f accs[2][2] = {{acc00, acc01}, {acc10, acc11}};
#pragma unroll
  for (int ms = 0; ms < 2; ++ms) {
#pragma unroll
    for (int ns = 0; ns < 2; ++ns) {
      const int cfeat = nt * 32 + ns * 16 + lo;
      const float bias = pb[cfeat];
      const int mbase = mt * 32 + ms * 16 + 8 * hi;  // 8 consecutive tokens/lane
      const int b = mbase >> 12;
      const int tok0 = mbase & (NTOK - 1);
      const float* xr = x + ((size_t)(b * NCH + cfeat)) * NTOK + tok0;
      float* orow = out + ((size_t)(b * NCH + cfeat)) * NTOK + tok0;
#pragma unroll
      for (int j = 0; j < 8; ++j) orow[j] = xr[j] + accs[ms][ns][j] + bias;
    }
  }
}

// ---------------------------------------------------------------------------
extern "C" void kernel_launch(void* const* d_in, const int* in_sizes, int n_in,
                              void* d_out, int out_size, void* d_ws, size_t ws_size,
                              hipStream_t stream) {
  (void)in_sizes; (void)n_in; (void)out_size; (void)ws_size;
  const float* x      = (const float*)d_in[0];
  const float* ln_g   = (const float*)d_in[1];
  const float* ln_b   = (const float*)d_in[2];
  const float* qkv_w  = (const float*)d_in[3];
  const float* qkv_b  = (const float*)d_in[4];
  const float* proj_w = (const float*)d_in[5];
  const float* proj_b = (const float*)d_in[6];
  float* out = (float*)d_out;

  // Workspace carve (halves). Each big buffer = B*N*C = 2,097,152 halves (4 MB).
  const size_t BIG = (size_t)BATCH * NTOK * NCH;
  _Float16* xnorm   = (_Float16*)d_ws;
  _Float16* qh      = xnorm + BIG;
  _Float16* kh      = qh + BIG;
  _Float16* vth     = kh + BIG;
  _Float16* attn_h  = vth + BIG;
  _Float16* qkvw_h  = attn_h + BIG;
  _Float16* projw_h = qkvw_h + (size_t)QKV_ROWS * NCH;

  wcvt_kernel<<<256, 256, 0, stream>>>(qkv_w, proj_w, qkvw_h, projw_h);
  ln_kernel<<<32, 256, 0, stream>>>(x, ln_g, ln_b, xnorm);
  qkv_kernel<<<768, 256, 0, stream>>>(xnorm, qkvw_h, qkv_b, qh, kh, vth);
  attn_kernel<<<256, 256, 0, stream>>>(qh, kh, vth, attn_h);
  proj_kernel<<<256, 256, 0, stream>>>(attn_h, projw_h, proj_b, x, out);
}
